// Attention_22600117911625
// MI455X (gfx1250) — compile-verified, matrix operand order checked by reading, not run on hardware
//
#include <hip/hip_runtime.h>

// ---------------------------------------------------------------------------
// MI455X (gfx1250) causal MHA with RoPE.
// bf16 V_WMMA_F32_16X16X32_BF16 for all GEMMs + wave32 flash attention.
// Global->LDS tile fills use the CDNA5 async pipe
// (global_load_async_to_lds_b128 / s_wait_asynccnt).
// ---------------------------------------------------------------------------

typedef __bf16 bf16_t;
typedef __attribute__((ext_vector_type(16))) __bf16 v16bf;
typedef __attribute__((ext_vector_type(8)))  __bf16 v8bf;
typedef __attribute__((ext_vector_type(8)))  float  v8f;

#define DEV __device__ __forceinline__

static constexpr int S_LEN = 4096;
static constexpr int DMODEL = 2048;
static constexpr int NHEADS = 16;
static constexpr int DHEAD = 128;
static constexpr int BATCH = 2;
static constexpr int ROWS = BATCH * S_LEN;   // 8192

DEV v8f wmma_bf16(v16bf a, v16bf b, v8f c) {
  // (neg_a, A, neg_b, B, c_mod, C, reuse_a, reuse_b)
  return __builtin_amdgcn_wmma_f32_16x16x32_bf16(false, a, false, b,
                                                 (short)0, c, false, false);
}

// Generic pointers to LDS carry the wave-relative LDS offset in the low 32
// bits.
DEV unsigned lds_off(const void* p) { return (unsigned)(uintptr_t)p; }

// Async DMA of 16 bytes per lane straight into LDS (no VGPR data path).
// Tracked by ASYNCcnt, not LOADcnt/DScnt.
DEV void async_copy_b128(unsigned lds_addr, const void* gptr) {
  asm volatile("global_load_async_to_lds_b128 %0, %1, off"
               :
               : "v"(lds_addr), "v"((unsigned long long)(uintptr_t)gptr)
               : "memory");
}
DEV void wait_async0() {
  asm volatile("s_wait_asynccnt 0x0" ::: "memory");
}

// Load one 16(wide) x 32(K) WMMA operand fragment from an LDS tile laid out
// [wide][k] row-major (leading dim `ld` bf16 elems).  Per the CDNA5 ISA 16-bit
// operand layout: lane = wide index (mod 16); lane-half selects K 0..7 vs 8..15
// in VGPR0-3 and +16 in VGPR4-7.  Two ds_load_b128 per fragment.
DEV v16bf load_frag(const bf16_t* tile, int ld, int wideOff, int kOff, int lane) {
  const int m = lane & 15;
  const int half = lane >> 4;
  const bf16_t* p = tile + (wideOff + m) * ld + kOff + half * 8;
  v8bf lo = *(const v8bf*)(p);
  v8bf hi = *(const v8bf*)(p + 16);
  return __builtin_shufflevector(lo, hi, 0, 1, 2, 3, 4, 5, 6, 7,
                                 8, 9, 10, 11, 12, 13, 14, 15);
}

// ---------------------------------------------------------------------------
// fp32 -> bf16 conversion
// ---------------------------------------------------------------------------
__global__ __launch_bounds__(256)
void f32_to_bf16(const float* __restrict__ in, bf16_t* __restrict__ out, int n) {
  const int i = blockIdx.x * 256 + threadIdx.x;
  if (i < n) out[i] = (bf16_t)in[i];
}

// ---------------------------------------------------------------------------
// C = A(MxK) * B(NxK)^T, A and B bf16 row-major with K contiguous.
// Workgroup tile 64(M) x 128(N); 8 waves arranged 2x4, each 32x32.
// MODE 0: scatter-store bf16 into [B,H,S,Dh]   (QKV projections)
// MODE 1: store fp32 row-major [row][DMODEL]    (output projection)
// ---------------------------------------------------------------------------
template <int MODE>
__global__ __launch_bounds__(256)
void gemm_wmma(const bf16_t* __restrict__ A, const bf16_t* __restrict__ Bw,
               void* __restrict__ Out) {
  constexpr int K = DMODEL;
  constexpr int BK = 32;
  __shared__ bf16_t As[64 * BK];    // 4 KB
  __shared__ bf16_t Bs[128 * BK];   // 8 KB

  const int tid = threadIdx.x;
  const int lane = tid & 31;
  const int wave = tid >> 5;
  const int wm = wave >> 2;                 // 0..1
  const int wn = wave & 3;                  // 0..3
  const int mBase = blockIdx.y * 64;
  const int nBase = blockIdx.x * 128;

  const int arow = tid >> 2, acol = (tid & 3) * 8;
  const int brow = tid >> 1, bcol = (tid & 1) * 16;

  // Per-thread fixed LDS slots and walking global pointers.
  const unsigned ldsA = lds_off(&As[arow * BK + acol]);
  const unsigned ldsB0 = lds_off(&Bs[brow * BK + bcol]);
  const unsigned ldsB1 = lds_off(&Bs[brow * BK + bcol + 8]);
  const bf16_t* gA = &A[(size_t)(mBase + arow) * K + acol];
  const bf16_t* gB = &Bw[(size_t)(nBase + brow) * K + bcol];

  v8f acc[2][2] = {};

  for (int k0 = 0; k0 < K; k0 += BK) {
    __syncthreads();
    async_copy_b128(ldsA, gA + k0);
    async_copy_b128(ldsB0, gB + k0);
    async_copy_b128(ldsB1, gB + k0 + 8);
    wait_async0();
    __syncthreads();

    v16bf a0 = load_frag(As, BK, wm * 32 + 0, 0, lane);
    v16bf a1 = load_frag(As, BK, wm * 32 + 16, 0, lane);
    v16bf b0 = load_frag(Bs, BK, wn * 32 + 0, 0, lane);
    v16bf b1 = load_frag(Bs, BK, wn * 32 + 16, 0, lane);
    acc[0][0] = wmma_bf16(a0, b0, acc[0][0]);
    acc[0][1] = wmma_bf16(a0, b1, acc[0][1]);
    acc[1][0] = wmma_bf16(a1, b0, acc[1][0]);
    acc[1][1] = wmma_bf16(a1, b1, acc[1][1]);
  }

  const int m = lane & 15, half = lane >> 4;
#pragma unroll
  for (int i = 0; i < 2; ++i)
#pragma unroll
    for (int j = 0; j < 2; ++j)
#pragma unroll
      for (int v = 0; v < 8; ++v) {
        const int r = mBase + wm * 32 + i * 16 + v + 8 * half;
        const int n = nBase + wn * 32 + j * 16 + m;
        const float val = acc[i][j][v];
        if (MODE == 0) {
          const int b = r >> 12, s = r & (S_LEN - 1);
          const int h = n >> 7, d = n & (DHEAD - 1);
          ((bf16_t*)Out)[((size_t)((b * NHEADS + h) * S_LEN + s)) * DHEAD + d] =
              (bf16_t)val;
        } else {
          ((float*)Out)[(size_t)r * DMODEL + n] = val;
        }
      }
}

// ---------------------------------------------------------------------------
// Interleaved-pair RoPE, in place on [B,H,S,Dh] bf16.  One thread per pair.
// ---------------------------------------------------------------------------
__global__ __launch_bounds__(256)
void rope_inplace(bf16_t* __restrict__ t, int npairs) {
  const int i = blockIdx.x * 256 + threadIdx.x;
  if (i >= npairs) return;
  const int d2 = i & 63;                       // pair index within head
  const int s = (i >> 6) & (S_LEN - 1);        // sequence position
  const float invf =
      __expf((float)(2 * d2) * (-9.210340371976184f / 128.0f));  // base^-2i/D
  const float ang = (float)s * invf;
  float sn, cs;
  __sincosf(ang, &sn, &cs);
  const float e0 = (float)t[2 * i];
  const float e1 = (float)t[2 * i + 1];
  t[2 * i] = (bf16_t)(e0 * cs - e1 * sn);
  t[2 * i + 1] = (bf16_t)(e1 * cs + e0 * sn);
}

// ---------------------------------------------------------------------------
// Flash attention, causal.  One workgroup = 128 query rows of one (b,h).
// 8 waves x 16 rows.  K tiles of 64 rows streamed through LDS via the async
// pipe; V is transposed into LDS [d][kv] so the PV B-operand is contiguous-K.
// Online softmax per wave with __shfl_xor reductions over the 16-lane column
// groups (wave32).
// ---------------------------------------------------------------------------
__global__ __launch_bounds__(256)
void flash_attn(const bf16_t* __restrict__ Q, const bf16_t* __restrict__ Kk,
                const bf16_t* __restrict__ V, bf16_t* __restrict__ Aout) {
  __shared__ bf16_t Qs[128 * 128];   // 32 KB
  __shared__ bf16_t Ks[64 * 128];    // 16 KB  [kv][d]
  __shared__ bf16_t Vts[128 * 64];   // 16 KB  [d][kv]
  __shared__ bf16_t Ps[8][16 * 64];  // 16 KB  per-wave P staging

  const int tid = threadIdx.x;
  const int lane = tid & 31;
  const int w = tid >> 5;
  const int m = lane & 15, half = lane >> 4;
  const int qt = blockIdx.x;
  const int bh = blockIdx.y;
  const int qBase = qt * 128;

  const bf16_t* Qg = Q + (size_t)bh * S_LEN * DHEAD;
  const bf16_t* Kg = Kk + (size_t)bh * S_LEN * DHEAD;
  const bf16_t* Vg = V + (size_t)bh * S_LEN * DHEAD;

#pragma unroll
  for (int c = 0; c < 8; ++c) {
    const int lin = (tid + c * 256) * 8;
    async_copy_b128(lds_off(&Qs[lin]), &Qg[(size_t)qBase * DHEAD + lin]);
  }
  wait_async0();
  __syncthreads();

  v16bf qf[4];
#pragma unroll
  for (int kk = 0; kk < 4; ++kk)
    qf[kk] = load_frag(Qs, DHEAD, w * 16, kk * 32, lane);

  float mrow[8], lrow[8];
  v8f o[8] = {};
#pragma unroll
  for (int v = 0; v < 8; ++v) { mrow[v] = -3.0e38f; lrow[v] = 0.0f; }

  const float scale = 0.08838834764831845f;  // 1/sqrt(128)
  const int qRowLast = qBase + w * 16 + 15;
  const int nkv = qt * 2 + 2;                // causal: kv <= qBase+127

  for (int kvt = 0; kvt < nkv; ++kvt) {
    const int kvBase = kvt * 64;
    __syncthreads();
#pragma unroll
    for (int c = 0; c < 4; ++c) {
      const int lin = (tid + c * 256) * 8;
      async_copy_b128(lds_off(&Ks[lin]), &Kg[(size_t)kvBase * DHEAD + lin]);
      const uint4 raw = *(const uint4*)&Vg[(size_t)kvBase * DHEAD + lin];
      const bf16_t* ev = (const bf16_t*)&raw;
      const int row = lin >> 7, col = lin & 127;
#pragma unroll
      for (int j = 0; j < 8; ++j) Vts[(col + j) * 64 + row] = ev[j];
    }
    wait_async0();
    __syncthreads();

    if (kvBase > qRowLast) continue;  // fully masked for this wave's rows

    // scores 16x64 = Q(16x128) . K(64x128)^T
    v8f sc[4] = {};
#pragma unroll
    for (int j = 0; j < 4; ++j)
#pragma unroll
      for (int kk = 0; kk < 4; ++kk)
        sc[j] = wmma_bf16(qf[kk], load_frag(Ks, DHEAD, j * 16, kk * 32, lane),
                          sc[j]);

    // scale + causal mask + row max
    float tmax[8];
#pragma unroll
    for (int v = 0; v < 8; ++v) tmax[v] = -3.0e38f;
    const int rowg0 = qBase + w * 16 + 8 * half;
#pragma unroll
    for (int j = 0; j < 4; ++j) {
      const int colg = kvBase + j * 16 + m;
#pragma unroll
      for (int v = 0; v < 8; ++v) {
        float sv = sc[j][v] * scale;
        sv = (colg <= rowg0 + v) ? sv : -3.0e38f;
        sc[j][v] = sv;
        tmax[v] = fmaxf(tmax[v], sv);
      }
    }
#pragma unroll
    for (int off = 1; off < 16; off <<= 1)
#pragma unroll
      for (int v = 0; v < 8; ++v)
        tmax[v] = fmaxf(tmax[v], __shfl_xor(tmax[v], off, 32));

    float alpha[8], rsum[8];
#pragma unroll
    for (int v = 0; v < 8; ++v) {
      const float mn = fmaxf(mrow[v], tmax[v]);
      alpha[v] = __expf(mrow[v] - mn);
      mrow[v] = mn;
      rsum[v] = 0.0f;
    }
    // P = exp(s - m); stage bf16 P for re-fragmentation as A operand
#pragma unroll
    for (int j = 0; j < 4; ++j)
#pragma unroll
      for (int v = 0; v < 8; ++v) {
        const float p = __expf(sc[j][v] - mrow[v]);
        rsum[v] += p;
        Ps[w][(v + 8 * half) * 64 + j * 16 + m] = (bf16_t)p;
      }
#pragma unroll
    for (int off = 1; off < 16; off <<= 1)
#pragma unroll
      for (int v = 0; v < 8; ++v) rsum[v] += __shfl_xor(rsum[v], off, 32);
#pragma unroll
    for (int v = 0; v < 8; ++v) lrow[v] = lrow[v] * alpha[v] + rsum[v];

    // rescale O then accumulate P(16x64) . V(64x128)
#pragma unroll
    for (int jd = 0; jd < 8; ++jd)
#pragma unroll
      for (int v = 0; v < 8; ++v) o[jd][v] = o[jd][v] * alpha[v];

    const v16bf pf0 = load_frag(&Ps[w][0], 64, 0, 0, lane);
    const v16bf pf1 = load_frag(&Ps[w][0], 64, 0, 32, lane);
#pragma unroll
    for (int jd = 0; jd < 8; ++jd) {
      o[jd] = wmma_bf16(pf0, load_frag(Vts, 64, jd * 16, 0, lane), o[jd]);
      o[jd] = wmma_bf16(pf1, load_frag(Vts, 64, jd * 16, 32, lane), o[jd]);
    }
  }

  const int b = bh >> 4, h = bh & 15;
  float inv[8];
#pragma unroll
  for (int v = 0; v < 8; ++v) inv[v] = 1.0f / lrow[v];
#pragma unroll
  for (int jd = 0; jd < 8; ++jd)
#pragma unroll
    for (int v = 0; v < 8; ++v) {
      const int rowg = qBase + w * 16 + v + 8 * half;
      const int col = jd * 16 + m;
      Aout[((size_t)(b * S_LEN + rowg)) * DMODEL + h * DHEAD + col] =
          (bf16_t)(o[jd][v] * inv[v]);
    }
}

// ---------------------------------------------------------------------------
extern "C" void kernel_launch(void* const* d_in, const int* /*in_sizes*/,
                              int /*n_in*/, void* d_out, int /*out_size*/,
                              void* d_ws, size_t /*ws_size*/,
                              hipStream_t stream) {
  const float* x = (const float*)d_in[0];
  const float* Wq = (const float*)d_in[1];
  const float* Wk = (const float*)d_in[2];
  const float* Wv = (const float*)d_in[3];
  const float* Wo = (const float*)d_in[4];

  const size_t nX = (size_t)ROWS * DMODEL;                 // 16.7M elems
  const size_t nW = (size_t)DMODEL * DMODEL;               // 4.2M elems
  const size_t nQKV = (size_t)BATCH * NHEADS * S_LEN * DHEAD;

  bf16_t* xb = (bf16_t*)d_ws;
  bf16_t* Wqb = xb + nX;
  bf16_t* Wkb = Wqb + nW;
  bf16_t* Wvb = Wkb + nW;
  bf16_t* Wob = Wvb + nW;
  bf16_t* Qb = Wob + nW;
  bf16_t* Kb = Qb + nQKV;
  bf16_t* Vb = Kb + nQKV;
  bf16_t* Ab = Vb + nQKV;

  f32_to_bf16<<<(int)((nX + 255) / 256), 256, 0, stream>>>(x, xb, (int)nX);
  f32_to_bf16<<<(int)((nW + 255) / 256), 256, 0, stream>>>(Wq, Wqb, (int)nW);
  f32_to_bf16<<<(int)((nW + 255) / 256), 256, 0, stream>>>(Wk, Wkb, (int)nW);
  f32_to_bf16<<<(int)((nW + 255) / 256), 256, 0, stream>>>(Wv, Wvb, (int)nW);
  f32_to_bf16<<<(int)((nW + 255) / 256), 256, 0, stream>>>(Wo, Wob, (int)nW);

  dim3 gGemm(DMODEL / 128, ROWS / 64);  // (16, 128)
  gemm_wmma<0><<<gGemm, 256, 0, stream>>>(xb, Wqb, Qb);
  gemm_wmma<0><<<gGemm, 256, 0, stream>>>(xb, Wkb, Kb);
  gemm_wmma<0><<<gGemm, 256, 0, stream>>>(xb, Wvb, Vb);

  const int npairs = BATCH * NHEADS * S_LEN * (DHEAD / 2);  // 8.4M
  rope_inplace<<<(npairs + 255) / 256, 256, 0, stream>>>(Qb, npairs);
  rope_inplace<<<(npairs + 255) / 256, 256, 0, stream>>>(Kb, npairs);

  dim3 gAttn(S_LEN / 128, BATCH * NHEADS);  // (32, 32)
  flash_attn<<<gAttn, 256, 0, stream>>>(Qb, Kb, Vb, Ab);

  gemm_wmma<1><<<gGemm, 256, 0, stream>>>(Ab, Wob, d_out);
}
